// EuclidDistance_Assign_Module_41540923687208
// MI455X (gfx1250) — matrix operand order repaired and checked by reading, not compiled, not run on hardware
//
#include <hip/hip_runtime.h>
#include <hip/hip_bf16.h>
#include <stdint.h>

// Problem constants (match reference)
#define NROWS 131072
#define DDIM  512
#define KCLUS 1024
#define ALPHA 32.0f
#define LNEPS 1e-5f

typedef __bf16 bf16;
typedef bf16  v16bf __attribute__((ext_vector_type(16)));
typedef bf16  v8bf  __attribute__((ext_vector_type(8)));
typedef float v8f   __attribute__((ext_vector_type(8)));

union V16U { v16bf v; v8bf h[2]; };

// ---- bf16 helpers (explicit RNE split; avoids relying on native __bf16 cvt) ----
static __device__ __forceinline__ bf16 f2bf(float f) {
    unsigned u = __float_as_uint(f);
    unsigned r = (u + 0x7FFFu + ((u >> 16) & 1u)) >> 16;   // round to nearest even
    unsigned short us = (unsigned short)r;
    return __builtin_bit_cast(bf16, us);
}
static __device__ __forceinline__ float bf2f(bf16 h) {
    unsigned short us = __builtin_bit_cast(unsigned short, h);
    return __uint_as_float(((unsigned)us) << 16);
}

// ---- WMMA wrapper: D = A(16x32 bf16) x B(32x16 bf16) + C(16x16 f32) ----
static __device__ __forceinline__ v8f wmma_bf16(v16bf a, v16bf b, v8f c) {
    return __builtin_amdgcn_wmma_f32_16x16x32_bf16(
        /*neg_a=*/false, a, /*neg_b=*/false, b,
        /*c_mod=*/(short)0, c, /*reuse_a=*/false, /*reuse_b=*/false);
}

// A-fragment (16x32, row-major source with row stride `stride`):
// lane<16: m=lane, elems0-15 = k+0..15 contiguous
// lane>=16: m=lane-16, elems0-15 = k+16..31 contiguous
static __device__ __forceinline__ v16bf load_a_frag(const bf16* base, int stride,
                                                    int kbase, int lane) {
    int m  = lane & 15;
    int hi = lane >> 4;
    const bf16* p = base + (size_t)m * stride + kbase + hi * 16;
    V16U u;
    u.h[0] = *(const v8bf*)(p);
    u.h[1] = *(const v8bf*)(p + 8);
    return u.v;
}

// B-fragment (32x16): lane<16: n=lane, k = kbase+0..15 contiguous;
// lane>=16: n=lane-16, k = kbase+16..31 contiguous.
// `p` must point at element (k = kbase + (lane>>4)*16, n = column for this lane).
static __device__ __forceinline__ v16bf load_b_frag(const bf16* p) {
    V16U u;
    u.h[0] = *(const v8bf*)(p);
    u.h[1] = *(const v8bf*)(p + 8);
    return u.v;
}

// ============================================================================
// Prep kernel: split cluster_center into bf16 hi/lo in two layouts + |c|^2
// ============================================================================
__global__ void prep_clusters(const float* __restrict__ C,
                              bf16* __restrict__ Chi, bf16* __restrict__ Clo,
                              bf16* __restrict__ CThi, bf16* __restrict__ CTlo,
                              float* __restrict__ c2) {
    int k = blockIdx.x;          // cluster 0..1023
    int t = threadIdx.x;         // 0..255
    __shared__ float red[256];
    float ss = 0.f;
    #pragma unroll
    for (int j = 0; j < 2; ++j) {
        int d = t + j * 256;
        float v = C[(size_t)k * DDIM + d];
        bf16 h = f2bf(v);
        bf16 l = f2bf(v - bf2f(h));
        Chi [(size_t)k * DDIM + d] = h;
        Clo [(size_t)k * DDIM + d] = l;
        CThi[(size_t)d * KCLUS + k] = h;
        CTlo[(size_t)d * KCLUS + k] = l;
        ss += v * v;
    }
    red[t] = ss;
    __syncthreads();
    for (int ofs = 128; ofs > 0; ofs >>= 1) {
        if (t < ofs) red[t] += red[t + ofs];
        __syncthreads();
    }
    if (t == 0) c2[k] = red[0];
}

// ============================================================================
// Fused kernel: 16 rows per block; LN -> cdist (WMMA) -> softmax -> rec (WMMA)
// ============================================================================
__global__ void fused_ln_dist_soft_rec(
        const float* __restrict__ x,
        const float* __restrict__ lnw,
        const float* __restrict__ lnb,
        const bf16* __restrict__ Chi, const bf16* __restrict__ Clo,
        const bf16* __restrict__ CThi, const bf16* __restrict__ CTlo,
        const float* __restrict__ c2,
        float* __restrict__ out_d,   // N x K
        float* __restrict__ out_s,   // N x K
        float* __restrict__ out_r) { // N x D
    extern __shared__ char smem[];
    bf16*     Ahi      = (bf16*)(smem);                 // 16 x 512 bf16 (16KB)
    bf16*     Alo      = (bf16*)(smem + 16 * 1024);     // 16 x 512 bf16 (16KB)
    bf16*     Shi      = (bf16*)(smem + 32 * 1024);     // 16 x 1024 bf16 (32KB)
    bf16*     Slo      = (bf16*)(smem + 64 * 1024);     // 16 x 1024 bf16 (32KB)
    unsigned* rowmin_u = (unsigned*)(smem + 96 * 1024);         // 16 u32
    float*    rowsum   = (float*)(smem + 96 * 1024 + 64);       // 16 f32
    float*    x2s      = (float*)(smem + 96 * 1024 + 128);      // 16 f32

    const int t    = threadIdx.x;     // 0..255
    const int lane = t & 31;
    const int w    = t >> 5;          // wave 0..7
    const int ln   = lane & 15;
    const int hi   = lane >> 4;
    const int row0 = blockIdx.x * 16;

    // ---------------- Phase 1: LayerNorm (16 threads per row) ----------------
    {
        int r  = t >> 4;      // row within block
        int c0 = t & 15;
        const float* xrow = x + (size_t)(row0 + r) * DDIM;
        float vals[32];
        float s = 0.f, s2 = 0.f;
        #pragma unroll
        for (int j = 0; j < 32; ++j) {
            float v = xrow[c0 + j * 16];
            vals[j] = v;
            s  += v;
            s2 += v * v;
        }
        // xor-reduce within 16-lane half-wave: all lanes get row totals
        #pragma unroll
        for (int m = 1; m < 16; m <<= 1) {
            s  += __shfl_xor(s,  m, 32);
            s2 += __shfl_xor(s2, m, 32);
        }
        float mean = s * (1.0f / DDIM);
        float var  = fmaxf(s2 * (1.0f / DDIM) - mean * mean, 0.f);
        float sc   = 1.0f / (sqrtf(var) + LNEPS);

        float xs2 = 0.f;
        #pragma unroll
        for (int j = 0; j < 32; ++j) {
            int   c  = c0 + j * 16;
            float xn = (vals[j] - mean) * sc * lnw[c] + lnb[c];
            xs2 += xn * xn;
            bf16 h = f2bf(xn);
            bf16 l = f2bf(xn - bf2f(h));
            Ahi[r * DDIM + c] = h;
            Alo[r * DDIM + c] = l;
        }
        #pragma unroll
        for (int m = 1; m < 16; m <<= 1) xs2 += __shfl_xor(xs2, m, 32);
        if (c0 == 0) x2s[r] = xs2;
        if (t < 16) { rowmin_u[t] = 0x7F800000u; rowsum[t] = 0.f; }
    }
    __syncthreads();

    // ---------- Phase 2: distances GEMM, wave w owns clusters [w*128, w*128+128) ----------
    v8f acc[8];
    #pragma unroll
    for (int s = 0; s < 8; ++s) acc[s] = (v8f){0.f,0.f,0.f,0.f,0.f,0.f,0.f,0.f};
    const int nb = w * 128;
    for (int kk = 0; kk < DDIM; kk += 32) {
        v16bf aH = load_a_frag(Ahi, DDIM, kk, lane);
        v16bf aL = load_a_frag(Alo, DDIM, kk, lane);
        #pragma unroll
        for (int s = 0; s < 8; ++s) {
            size_t boff = (size_t)(nb + s * 16 + ln) * DDIM + kk + hi * 16;
            v16bf bH = load_b_frag(Chi + boff);
            v16bf bL = load_b_frag(Clo + boff);
            acc[s] = wmma_bf16(aH, bH, acc[s]);   // hi*hi
            acc[s] = wmma_bf16(aH, bL, acc[s]);   // hi*lo
            acc[s] = wmma_bf16(aL, bH, acc[s]);   // lo*hi
        }
    }

    // ---------- Phase 3a: d = sqrt(max(|x|^2 + |c|^2 - 2 dot, 0)); row-min ----------
    #pragma unroll
    for (int v = 0; v < 8; ++v) {
        int   m    = v + hi * 8;
        float x2m  = x2s[m];
        float dmin = 3.4e38f;
        #pragma unroll
        for (int s = 0; s < 8; ++s) {
            int   n   = nb + s * 16 + ln;
            float dot = acc[s][v];
            float d2  = x2m + c2[n] - 2.0f * dot;
            float d   = sqrtf(fmaxf(d2, 0.f));
            out_d[(size_t)(row0 + m) * KCLUS + n] = d;
            acc[s][v] = d;
            dmin = fminf(dmin, d);
        }
        #pragma unroll
        for (int msk = 1; msk < 16; msk <<= 1)
            dmin = fminf(dmin, __shfl_xor(dmin, msk, 32));
        if (ln == 0) atomicMin(&rowmin_u[m], __float_as_uint(dmin)); // d>=0: bit-order ok
    }
    __syncthreads();

    // ---------- Phase 3b: e = exp(-alpha*(d - dmin)); row-sum ----------
    #pragma unroll
    for (int v = 0; v < 8; ++v) {
        int   m  = v + hi * 8;
        float dm = __uint_as_float(rowmin_u[m]);
        float es = 0.f;
        #pragma unroll
        for (int s = 0; s < 8; ++s) {
            float e = __expf(-ALPHA * (acc[s][v] - dm));
            acc[s][v] = e;
            es += e;
        }
        #pragma unroll
        for (int msk = 1; msk < 16; msk <<= 1) es += __shfl_xor(es, msk, 32);
        if (ln == 0) atomicAdd(&rowsum[m], es);
    }
    __syncthreads();

    // ---------- Phase 3c: soft = e / sum; write out + bf16 split into LDS ----------
    #pragma unroll
    for (int v = 0; v < 8; ++v) {
        int   m   = v + hi * 8;
        float inv = 1.0f / rowsum[m];
        #pragma unroll
        for (int s = 0; s < 8; ++s) {
            int   n = nb + s * 16 + ln;
            float p = acc[s][v] * inv;
            out_s[(size_t)(row0 + m) * KCLUS + n] = p;
            bf16 h = f2bf(p);
            bf16 l = f2bf(p - bf2f(h));
            Shi[m * KCLUS + n] = h;
            Slo[m * KCLUS + n] = l;
        }
    }
    __syncthreads();

    // ---------- Phase 4: x_rec = soft @ C; wave w owns d-columns [w*64, w*64+64) ----------
    v8f acc2[4];
    #pragma unroll
    for (int s = 0; s < 4; ++s) acc2[s] = (v8f){0.f,0.f,0.f,0.f,0.f,0.f,0.f,0.f};
    const int db = w * 64;
    for (int kc = 0; kc < KCLUS; kc += 32) {
        v16bf aH = load_a_frag(Shi, KCLUS, kc, lane);
        v16bf aL = load_a_frag(Slo, KCLUS, kc, lane);
        #pragma unroll
        for (int s = 0; s < 4; ++s) {
            size_t boff = (size_t)(db + s * 16 + ln) * KCLUS + kc + hi * 16;
            v16bf bH = load_b_frag(CThi + boff);
            v16bf bL = load_b_frag(CTlo + boff);
            acc2[s] = wmma_bf16(aH, bH, acc2[s]);
            acc2[s] = wmma_bf16(aH, bL, acc2[s]);
            acc2[s] = wmma_bf16(aL, bH, acc2[s]);
        }
    }
    #pragma unroll
    for (int s = 0; s < 4; ++s) {
        int nd = db + s * 16 + ln;
        #pragma unroll
        for (int v = 0; v < 8; ++v) {
            int m = v + hi * 8;
            out_r[(size_t)(row0 + m) * DDIM + nd] = acc2[s][v];
        }
    }
}

// ============================================================================
extern "C" void kernel_launch(void* const* d_in, const int* in_sizes, int n_in,
                              void* d_out, int out_size, void* d_ws, size_t ws_size,
                              hipStream_t stream) {
    (void)in_sizes; (void)n_in; (void)out_size; (void)ws_size;

    const float* x   = (const float*)d_in[0];   // N x D
    const float* lnw = (const float*)d_in[1];   // D
    const float* lnb = (const float*)d_in[2];   // D
    const float* C   = (const float*)d_in[3];   // K x D

    float* out   = (float*)d_out;
    float* out_d = out;                                   // N*K
    float* out_s = out + (size_t)NROWS * KCLUS;           // N*K
    float* out_r = out + 2 * (size_t)NROWS * KCLUS;       // N*D

    // Workspace layout (needs ~4.2 MB)
    float* c2  = (float*)d_ws;                                        // K f32
    bf16* Chi  = (bf16*)((char*)d_ws + 4096);                         // K*D
    bf16* Clo  = Chi  + (size_t)KCLUS * DDIM;                         // K*D
    bf16* CThi = Clo  + (size_t)KCLUS * DDIM;                         // D*K
    bf16* CTlo = CThi + (size_t)DDIM * KCLUS;                         // D*K

    prep_clusters<<<KCLUS, 256, 0, stream>>>(C, Chi, Clo, CThi, CTlo, c2);

    const int blocks = NROWS / 16;                // 8192
    const size_t shmem = 96 * 1024 + 192;         // Ahi/Alo + Shi/Slo + stats
    fused_ln_dist_soft_rec<<<blocks, 256, shmem, stream>>>(
        x, lnw, lnb, Chi, Clo, CThi, CTlo, c2, out_d, out_s, out_r);
}
